// HybridQuantumNet_30657476559155
// MI455X (gfx1250) — matrix operand order repaired and checked by reading, not compiled, not run on hardware
//
#include <hip/hip_runtime.h>

typedef __attribute__((ext_vector_type(16))) _Float16 v16h;
typedef __attribute__((ext_vector_type(8)))  float    v8f;

#define IN   64
#define HID  64
#define NQ   4
#define QDIM 16
#define NCOEF 81

__device__ __forceinline__ float fast_rcp(float v) { return __builtin_amdgcn_rcpf(v); }
__device__ __forceinline__ float silu_f(float v) {
    return v * fast_rcp(1.0f + __expf(-v));       // v * sigmoid(v), v_rcp not v_div chains
}
__device__ __forceinline__ float fast_tanh(float v) {
    return 1.0f - 2.0f * fast_rcp(__expf(2.0f * v) + 1.0f);
}

// ---------------------------------------------------------------------------
// One-time setup: collapse the fixed part of the circuit (Rot layers + CNOT
// rings) into an 81-coefficient multilinear form over {I, Z, Y}^4.
//   out_q(row) = sum_k coef[k] * prod_w g_w[k_w],  g_w = {1, cos a_w, -sin a_w}
// ---------------------------------------------------------------------------
__global__ __launch_bounds__(32)
void hqnet_setup_kernel(const float* __restrict__ qw,
                        const float* __restrict__ W3,
                        float* __restrict__ cf)
{
    __shared__ float Mr[QDIM][QDIM], Mi[QDIM][QDIM];
    __shared__ float Hr[QDIM][QDIM], Hi[QDIM][QDIM];
    const int t = threadIdx.x;

    // Column t of M: apply the fixed gate sequence to basis state |t>
    if (t < QDIM) {
        float sr[QDIM], si[QDIM];
        for (int i = 0; i < QDIM; ++i) { sr[i] = 0.f; si[i] = 0.f; }
        sr[t] = 1.f;

        for (int l = 0; l < 2; ++l) {
            for (int w = 0; w < NQ; ++w) {
                const float* q = qw + (l * NQ + w) * 3;
                float phi = q[0], th = q[1], om = q[2];
                float cth = __cosf(0.5f * th), sth = __sinf(0.5f * th);
                float ap = 0.5f * (phi + om), am = 0.5f * (phi - om);
                float cap = __cosf(ap), sap = __sinf(ap);
                float cam = __cosf(am), sam = __sinf(am);
                float u00r =  cap * cth, u00i = -sap * cth;
                float u01r = -cam * sth, u01i = -sam * sth;
                float u10r =  cam * sth, u10i = -sam * sth;
                float u11r =  cap * cth, u11i =  sap * cth;
                const int tb = 1 << (3 - w);
                for (int i = 0; i < QDIM; ++i) {
                    if (i & tb) continue;
                    int j = i | tb;
                    float r0 = sr[i], i0 = si[i], r1 = sr[j], i1 = si[j];
                    sr[i] = u00r * r0 - u00i * i0 + u01r * r1 - u01i * i1;
                    si[i] = u00r * i0 + u00i * r0 + u01r * i1 + u01i * r1;
                    sr[j] = u10r * r0 - u10i * i0 + u11r * r1 - u11i * i1;
                    si[j] = u10r * i0 + u10i * r0 + u11r * i1 + u11i * r1;
                }
            }
            const int rr = (l % (NQ - 1)) + 1;
            for (int w = 0; w < NQ; ++w) {
                const int cb = 1 << (3 - w);
                const int tb = 1 << (3 - ((w + rr) & 3));
                for (int i = 0; i < QDIM; ++i) {
                    if ((i & cb) && !(i & tb)) {
                        int j = i | tb;
                        float tr = sr[i]; sr[i] = sr[j]; sr[j] = tr;
                        float ti = si[i]; si[i] = si[j]; si[j] = ti;
                    }
                }
            }
        }
        for (int k = 0; k < QDIM; ++k) { Mr[k][t] = sr[k]; Mi[k][t] = si[k]; }
    }
    __syncthreads();

    // H = M^dagger * diag(d) * M,  d_a = sum_w W3[w] * (1 - 2*bit(3-w, a))
    const float w30 = W3[0], w31 = W3[1], w32 = W3[2], w33 = W3[3];
    for (int e = t; e < QDIM * QDIM; e += 32) {
        const int i = e >> 4, j = e & 15;
        float hr = 0.f, hi = 0.f;
        for (int k = 0; k < QDIM; ++k) {
            float d = (((k >> 3) & 1) ? -w30 : w30) + (((k >> 2) & 1) ? -w31 : w31)
                    + (((k >> 1) & 1) ? -w32 : w32) + (((k >> 0) & 1) ? -w33 : w33);
            float ar = Mr[k][i], ai = Mi[k][i], br = Mr[k][j], bi = Mi[k][j];
            hr += d * (ar * br + ai * bi);     // conj(M_ki) * M_kj
            hi += d * (ar * bi - ai * br);
        }
        Hr[i][j] = hr; Hi[i][j] = hi;
    }
    __syncthreads();

    // coef_k = (1/16) * Tr[H * (P_k0 x P_k1 x P_k2 x P_k3)],  P in {I,Z,Y}
    for (int k = t; k < NCOEF; k += 32) {
        const int kd[4] = { k / 27, (k / 9) % 3, (k / 3) % 3, k % 3 };
        int ym = 0, zm = 0, nY = 0;
        for (int w = 0; w < NQ; ++w) {
            const int bit = 1 << (3 - w);
            if      (kd[w] == 1) zm |= bit;
            else if (kd[w] == 2) { ym |= bit; ++nY; }
        }
        float sre = 0.f, sim = 0.f;
        for (int a = 0; a < QDIM; ++a) {
            const int b = a ^ ym;
            const float sg = (__popc(a & (zm | ym)) & 1) ? -1.f : 1.f;
            sre += sg * Hr[a][b];
            sim += sg * Hi[a][b];
        }
        float cr, ci;   // i^nY
        switch (nY & 3) {
            case 0:  cr =  1.f; ci =  0.f; break;
            case 1:  cr =  0.f; ci =  1.f; break;
            case 2:  cr = -1.f; ci =  0.f; break;
            default: cr =  0.f; ci = -1.f; break;
        }
        cf[k] = (cr * sre - ci * sim) * (1.f / 16.f);
    }
}

// ---------------------------------------------------------------------------
// Fused main kernel: x -> WMMA GEMM1 + SiLU -> WMMA GEMM2 + tanh -> 81-term
// multilinear quantum expectation -> out.  One HBM read of x, one B-float write.
// ---------------------------------------------------------------------------
__global__ __launch_bounds__(256)
void hqnet_fused_kernel(const float* __restrict__ x,
                        const float* __restrict__ W1,
                        const float* __restrict__ b1,
                        const float* __restrict__ W2,
                        const float* __restrict__ b2,
                        const float* __restrict__ b3,
                        const float* __restrict__ cf,
                        float* __restrict__ out,
                        int totalRows)
{
    __shared__ _Float16 sXH[256 * IN];       // 32 KB: x tile f16, reused as h tile f16
    __shared__ _Float16 sW1[HID * IN];       //  8 KB: W1 f16 [n][k]
    __shared__ _Float16 sW2h[16 * IN];       //  2 KB: W2 padded to 16 rows, f16 [n][k]
    __shared__ float    sB1[HID];
    __shared__ float    sB2[16];
    __shared__ float    sAng[256 * NQ];      //  4 KB: per-row angles

    const int tid     = threadIdx.x;
    const int rowBase = blockIdx.x * 256;

    // ---------------- Phase 0: stage operands (vectorized, coalesced) ----------------
    for (int i = tid * 4; i < HID * IN; i += 256 * 4) {
        float4 w = *(const float4*)(W1 + i);
        union { _Float16 h[4]; unsigned long long u; } pk;
        pk.h[0] = (_Float16)w.x; pk.h[1] = (_Float16)w.y;
        pk.h[2] = (_Float16)w.z; pk.h[3] = (_Float16)w.w;
        *(unsigned long long*)(sW1 + i) = pk.u;
    }
    for (int i = tid * 4; i < 256 * IN; i += 256 * 4) {
        const int gi = rowBase * IN + i;
        union { _Float16 h[4]; unsigned long long u; } pk;
        if (gi + 3 < totalRows * IN) {
            float4 v = *(const float4*)(x + (size_t)gi);
            pk.h[0] = (_Float16)v.x; pk.h[1] = (_Float16)v.y;
            pk.h[2] = (_Float16)v.z; pk.h[3] = (_Float16)v.w;
        } else {
            pk.u = 0ull;
        }
        *(unsigned long long*)(sXH + i) = pk.u;
    }
    if (tid < 16 * IN) {                      // W2 padded 4x64 -> 16x64
        const int n = tid >> 6, k = tid & 63;
        sW2h[tid] = (_Float16)((n < NQ) ? W2[n * IN + k] : 0.0f);
    }
    if (tid < HID) sB1[tid] = b1[tid];
    if (tid < 16)  sB2[tid] = (tid < NQ) ? b2[tid] : 0.0f;
    __syncthreads();

    // ---------------- Phase 1: GEMM1 (WMMA) + bias + SiLU, h back to LDS ----------------
    const int wave = tid >> 5;
    const int lane = tid & 31;
    const int lhi  = lane >> 4;
    const int l16  = lane & 15;

    const unsigned* sXHu  = (const unsigned*)sXH;
    const unsigned* sW1u  = (const unsigned*)sW1;
    const unsigned* sW2hu = (const unsigned*)sW2h;

    #pragma unroll
    for (int mtile = 0; mtile < 2; ++mtile) {
        const int rowA = wave * 32 + mtile * 16 + l16;

        v16h afrag[2];
        #pragma unroll
        for (int ks = 0; ks < 2; ++ks) {
            union { v16h v; unsigned u[8]; } A;
            #pragma unroll
            for (int vv = 0; vv < 8; ++vv) {
                int kk = ks * 32 + ((vv < 4) ? 0 : 16) + lhi * 8 + 2 * (vv & 3);
                A.u[vv] = sXHu[(rowA * IN + kk) >> 1];
            }
            afrag[ks] = A.v;
        }

        v8f acc[4] = {};
        #pragma unroll
        for (int nt = 0; nt < 4; ++nt) {
            const int n = nt * 16 + l16;
            #pragma unroll
            for (int ks = 0; ks < 2; ++ks) {
                union { v16h v; unsigned u[8]; } Bf;
                #pragma unroll
                for (int vv = 0; vv < 8; ++vv) {
                    int kk = ks * 32 + ((vv < 4) ? 0 : 16) + lhi * 8 + 2 * (vv & 3);
                    Bf.u[vv] = sW1u[(n * IN + kk) >> 1];
                }
                acc[nt] = __builtin_amdgcn_wmma_f32_16x16x32_f16(
                    false, afrag[ks], false, Bf.v, (short)0, acc[nt], false, false);
            }
        }

        #pragma unroll
        for (int nt = 0; nt < 4; ++nt) {
            const float bias = sB1[nt * 16 + l16];
            #pragma unroll
            for (int r = 0; r < 8; ++r) {
                float v = silu_f(acc[nt][r] + bias);
                int m = wave * 32 + mtile * 16 + lhi * 8 + r;
                sXH[m * IN + nt * 16 + l16] = (_Float16)v;
            }
        }
    }
    asm volatile("s_wait_dscnt 0x0" ::: "memory");

    // ---------------- Phase 1b: GEMM2 (WMMA, N padded to 16) + tanh -> sAng ----------------
    #pragma unroll
    for (int mtile = 0; mtile < 2; ++mtile) {
        const int rowA = wave * 32 + mtile * 16 + l16;

        v8f acc2 = {};
        #pragma unroll
        for (int ks = 0; ks < 2; ++ks) {
            union { v16h v; unsigned u[8]; } A;
            #pragma unroll
            for (int vv = 0; vv < 8; ++vv) {
                int kk = ks * 32 + ((vv < 4) ? 0 : 16) + lhi * 8 + 2 * (vv & 3);
                A.u[vv] = sXHu[(rowA * IN + kk) >> 1];
            }
            union { v16h v; unsigned u[8]; } Bf;
            #pragma unroll
            for (int vv = 0; vv < 8; ++vv) {
                int kk = ks * 32 + ((vv < 4) ? 0 : 16) + lhi * 8 + 2 * (vv & 3);
                Bf.u[vv] = sW2hu[(l16 * IN + kk) >> 1];
            }
            acc2 = __builtin_amdgcn_wmma_f32_16x16x32_f16(
                false, A.v, false, Bf.v, (short)0, acc2, false, false);
        }

        if (l16 < NQ) {
            const float bias2 = sB2[l16];
            #pragma unroll
            for (int r = 0; r < 8; ++r) {
                float a = fast_tanh(acc2[r] + bias2);
                int m = wave * 32 + mtile * 16 + lhi * 8 + r;
                sAng[m * NQ + l16] = a;
            }
        }
    }
    asm volatile("s_wait_dscnt 0x0" ::: "memory");

    // ---------------- Phase 2: 81-term multilinear quantum expectation ----------------
    const int lrow = wave * 32 + lane;
    const int grow = rowBase + lrow;

    const float a0 = sAng[lrow * NQ + 0];
    const float a1 = sAng[lrow * NQ + 1];
    const float a2 = sAng[lrow * NQ + 2];
    const float a3 = sAng[lrow * NQ + 3];

    const float g0[3] = { 1.f, __cosf(a0), -__sinf(a0) };
    const float g1[3] = { 1.f, __cosf(a1), -__sinf(a1) };
    const float g2[3] = { 1.f, __cosf(a2), -__sinf(a2) };
    const float g3[3] = { 1.f, __cosf(a3), -__sinf(a3) };

    float r0 = 0.f;
    #pragma unroll
    for (int k0 = 0; k0 < 3; ++k0) {
        float r1 = 0.f;
        #pragma unroll
        for (int k1 = 0; k1 < 3; ++k1) {
            float r2 = 0.f;
            #pragma unroll
            for (int k2 = 0; k2 < 3; ++k2) {
                const float* c3 = cf + (k0 * 27 + k1 * 9 + k2 * 3);
                float r3 = c3[0] + c3[1] * g3[1] + c3[2] * g3[2];   // cf uniform -> s_loads
                r2 += r3 * g2[k2];
            }
            r1 += r2 * g1[k1];
        }
        r0 += r1 * g0[k0];
    }

    if (grow < totalRows) out[grow] = r0 + b3[0];
}

extern "C" void kernel_launch(void* const* d_in, const int* in_sizes, int n_in,
                              void* d_out, int out_size, void* d_ws, size_t ws_size,
                              hipStream_t stream) {
    const float* x  = (const float*)d_in[0];
    const float* W1 = (const float*)d_in[1];
    const float* b1 = (const float*)d_in[2];
    const float* W2 = (const float*)d_in[3];
    const float* b2 = (const float*)d_in[4];
    const float* qw = (const float*)d_in[5];
    const float* W3 = (const float*)d_in[6];
    const float* b3 = (const float*)d_in[7];
    float* out = (float*)d_out;
    float* cf  = (float*)d_ws;                    // 81 floats of scratch

    const int totalRows = in_sizes[0] / IN;       // 262144
    const int grid = (totalRows + 255) / 256;     // 1024 blocks of 256 rows

    hqnet_setup_kernel<<<1, 32, 0, stream>>>(qw, W3, cf);
    hqnet_fused_kernel<<<grid, 256, 0, stream>>>(x, W1, b1, W2, b2, b3, cf,
                                                 out, totalRows);
}
